// GCNSpanEmbedder_43430709297357
// MI455X (gfx1250) — matrix operand order repaired and checked by reading, not compiled, not run on hardware
//
#include <hip/hip_runtime.h>
#include <hip/hip_bf16.h>

typedef __bf16 bf16;
typedef __attribute__((ext_vector_type(16))) __bf16 v16bf;
typedef __attribute__((ext_vector_type(8)))  float  v8f;
typedef __attribute__((ext_vector_type(2)))  float  f32x2;

union FragBF { unsigned int u[8]; v16bf v; };

__device__ __forceinline__ unsigned int pack_bf16(float a, float b) {
  union { bf16 h[2]; unsigned int u; } r;
  r.h[0] = (bf16)a; r.h[1] = (bf16)b;   // lowers to v_cvt_pk_bf16_f32
  return r.u;
}
// load 2 consecutive elements starting at an even index; convert/pack to bf16x2
__device__ __forceinline__ unsigned int load2_cvt(const float* p) {
  f32x2 v = *(const f32x2*)p;           // 8B-aligned (ld even, k even)
  return pack_bf16(v.x, v.y);
}
__device__ __forceinline__ unsigned int load2_cvt(const bf16* p) {
  return *(const unsigned int*)p;       // already bf16 pair, 4B-aligned
}

// ---- tile staging: 128x32 A / 32x128 B as packed K-pairs --------------------
template<bool G, typename TA>
__device__ __forceinline__ void stage_A(unsigned int (*__restrict__ sAu)[18],
                                        const TA* __restrict__ A, long lda,
                                        int m0, int k0, int K, int tid)
{
  #pragma unroll
  for (int i = 0; i < 8; ++i) {         // 2048 pairs / 256 threads
    int idx = tid + i * 256;
    int k2 = idx & 15, m = idx >> 4;    // coalesced along K
    int gk = k0 + 2 * k2;
    const TA* p = A + (long)(m0 + m) * lda + gk;
    unsigned int u;
    if (G) {
      float a = (gk     < K) ? (float)p[0] : 0.0f;
      float b = (gk + 1 < K) ? (float)p[1] : 0.0f;
      u = pack_bf16(a, b);
    } else {
      u = load2_cvt(p);
    }
    sAu[m][k2] = u;
  }
}

template<bool G, bool BTRANS, typename TB>
__device__ __forceinline__ void stage_B(unsigned int (*__restrict__ sBu)[128],
                                        const TB* __restrict__ Bm, long ldb,
                                        int n0, int k0, int K, int tid)
{
  #pragma unroll
  for (int i = 0; i < 8; ++i) {
    int idx = tid + i * 256;
    unsigned int u;
    int k2, n;
    if (!BTRANS) {
      k2 = idx >> 7; n = idx & 127;     // coalesced along N
      int gk = k0 + 2 * k2;
      const TB* p = Bm + (long)gk * ldb + (n0 + n);
      float a, b;
      if (G) {
        a = (gk     < K) ? (float)p[0]   : 0.0f;
        b = (gk + 1 < K) ? (float)p[ldb] : 0.0f;
      } else {
        a = (float)p[0]; b = (float)p[ldb];
      }
      u = pack_bf16(a, b);
    } else {
      k2 = idx & 15; n = idx >> 4;      // coalesced along K
      int gk = k0 + 2 * k2;
      const TB* p = Bm + (long)(n0 + n) * ldb + gk;
      if (G) {
        float a = (gk     < K) ? (float)p[0] : 0.0f;
        float b = (gk + 1 < K) ? (float)p[1] : 0.0f;
        u = pack_bf16(a, b);
      } else {
        u = load2_cvt(p);
      }
    }
    sBu[k2][n] = u;
  }
}

// ---------------------------------------------------------------------------
// Generic bf16-WMMA GEMM: C[M,N] = A[M,K] * B[K,N] (+bias[N]) (+relu)
//  - A/B sources f32 or bf16, converted while staging to LDS
//  - BTRANS: B stored [N,K] row-major (weights)
//  - dual-stride batching: z -> (i0 = z % inner, i1 = z / inner)
//  - M, N multiples of 128; K zero-padded (guard only in peeled tail tile)
// 128x128 macro tile, 8 waves * (2x4) wmma_f32_16x16x32_bf16, LDS double-buffer
// ---------------------------------------------------------------------------
template<typename TA, typename TB, bool BTRANS, bool RELU, typename TC>
__global__ void __launch_bounds__(256) gemm_wmma_bf16(
    const TA* __restrict__ A, const TB* __restrict__ Bm,
    const float* __restrict__ bias, TC* __restrict__ C,
    int M, int N, int K,
    long lda, long ldb, long ldc,
    int inner,
    long sa0, long sa1, long sb0, long sb1,
    long sc0, long sc1, long sbias0, long sbias1)
{
  __shared__ unsigned int sAu[2][128][18];  // [buf][m][k2], stride 18 -> 16-way banks
  __shared__ unsigned int sBu[2][16][128];  // [buf][k2][n]

  const int z  = blockIdx.z;
  const int i0 = z % inner, i1 = z / inner;
  A  += (long)i0 * sa0 + (long)i1 * sa1;
  Bm += (long)i0 * sb0 + (long)i1 * sb1;
  C  += (long)i0 * sc0 + (long)i1 * sc1;
  if (bias) bias += (long)i0 * sbias0 + (long)i1 * sbias1;

  const int tid  = threadIdx.x;
  const int lane = tid & 31;
  const int half = lane >> 4;
  const int l16  = lane & 15;
  const int wave = tid >> 5;
  const int wmb  = (wave & 3) * 32;
  const int wnb  = (wave >> 2) * 64;

  const int m0 = blockIdx.x * 128;
  const int n0 = blockIdx.y * 128;

  v8f acc[2][4];
  const v8f vzero = {0.f,0.f,0.f,0.f,0.f,0.f,0.f,0.f};
  #pragma unroll
  for (int i = 0; i < 2; ++i)
    #pragma unroll
    for (int j = 0; j < 4; ++j) acc[i][j] = vzero;

  const int  nk   = (K + 31) >> 5;
  const bool tail = (K & 31) != 0;

  // prologue: stage tile 0
  if (tail && nk == 1) {
    stage_A<true >(sAu[0], A, lda, m0, 0, K, tid);
    stage_B<true , BTRANS>(sBu[0], Bm, ldb, n0, 0, K, tid);
  } else {
    stage_A<false>(sAu[0], A, lda, m0, 0, K, tid);
    stage_B<false, BTRANS>(sBu[0], Bm, ldb, n0, 0, K, tid);
  }

  int buf = 0;
  for (int ks = 0; ks < nk; ++ks) {
    __syncthreads();
    // stage next tile into the other buffer while computing this one
    const int nxt = ks + 1;
    if (nxt < nk) {
      const int k0 = nxt << 5;
      if (tail && nxt == nk - 1) {
        stage_A<true >(sAu[buf ^ 1], A, lda, m0, k0, K, tid);
        stage_B<true , BTRANS>(sBu[buf ^ 1], Bm, ldb, n0, k0, K, tid);
      } else {
        stage_A<false>(sAu[buf ^ 1], A, lda, m0, k0, K, tid);
        stage_B<false, BTRANS>(sBu[buf ^ 1], Bm, ldb, n0, k0, K, tid);
      }
    }

    // fragments in contiguous K order (k = 16*half + s), same for A and B
    FragBF afr[2], bfr[4];
    #pragma unroll
    for (int mi = 0; mi < 2; ++mi) {
      int m = wmb + mi * 16 + l16;            // A: lane = row M
      #pragma unroll
      for (int j = 0; j < 8; ++j) {
        int k2 = j + 8 * half;
        afr[mi].u[j] = sAu[buf][m][k2];
      }
    }
    #pragma unroll
    for (int ni = 0; ni < 4; ++ni) {
      int n = wnb + ni * 16 + l16;            // B: lane = col N
      #pragma unroll
      for (int j = 0; j < 8; ++j)             // VGPR j: K = 16*half + 2j,2j+1
        bfr[ni].u[j] = sBu[buf][j + 8 * half][n];
    }
    #pragma unroll
    for (int mi = 0; mi < 2; ++mi)
      #pragma unroll
      for (int ni = 0; ni < 4; ++ni)
        acc[mi][ni] = __builtin_amdgcn_wmma_f32_16x16x32_bf16(
            false, afr[mi].v, false, bfr[ni].v, (short)0, acc[mi][ni],
            false, false);
    buf ^= 1;
  }

  // epilogue: bias / relu / store
  #pragma unroll
  for (int mi = 0; mi < 2; ++mi)
    #pragma unroll
    for (int ni = 0; ni < 4; ++ni) {
      int gn = n0 + wnb + ni * 16 + l16;
      float bv = bias ? bias[gn] : 0.0f;
      #pragma unroll
      for (int j = 0; j < 8; ++j) {           // C: VGPR j -> row 8*half + j
        int gm = m0 + wmb + mi * 16 + half * 8 + j;
        float v = acc[mi][ni][j] + bv;
        if (RELU) v = fmaxf(v, 0.0f);
        C[(long)gm * ldc + gn] = static_cast<TC>(v);
      }
    }
}

// ---------------------------------------------------------------------------
// sc[r] = dot(h1[r, 0:128], W2) + b2     (h1 already relu(..)+b1 from GEMM)
// ---------------------------------------------------------------------------
__global__ void sc_dot_kernel(const float* __restrict__ h1,
                              const float* __restrict__ W2,
                              const float* __restrict__ b2,
                              float* __restrict__ sc, int rows)
{
  int r = blockIdx.x * blockDim.x + threadIdx.x;
  if (r >= rows) return;
  const float* row = h1 + (size_t)r * 128;
  float s = 0.f;
  #pragma unroll 8
  for (int k = 0; k < 128; ++k) s += row[k] * W2[k];
  sc[r] = s + b2[0];
}

// ---------------------------------------------------------------------------
// Per span n: masked softmax over M=16, attention-weighted sum of cont, and
// assembly of vec[n, 0:2354] as bf16:
//   [start(768) | end(768) | wsum(768) | width_emb(20) | type_emb(30)]
// ---------------------------------------------------------------------------
__global__ void softmax_vec_kernel(const float* __restrict__ sc,
                                   const float* __restrict__ cont,
                                   const float* __restrict__ startv,
                                   const float* __restrict__ endv,
                                   const float* __restrict__ wemb,
                                   const float* __restrict__ temb,
                                   const int* __restrict__ width,
                                   const int* __restrict__ tlab,
                                   bf16* __restrict__ vec)
{
  const int n = blockIdx.x;              // 0..2047
  __shared__ float smem[16];
  int w  = width[n];
  int wm = (w > 1) ? w : 1;
  if (threadIdx.x < 16) {
    int m = threadIdx.x;
    float v = sc[n * 16 + m];
    v = (m < wm) ? v : 0.0f;             // sc *= mask
    v = (v != 0.0f) ? v : -9e9f;         // where(sc != 0, sc, NEG)
    smem[m] = v;
  }
  __syncthreads();
  float a[16];
  float mx = -3.4e38f;
  #pragma unroll
  for (int m = 0; m < 16; ++m) { a[m] = smem[m]; mx = fmaxf(mx, a[m]); }
  float sum = 0.f;
  #pragma unroll
  for (int m = 0; m < 16; ++m) { a[m] = __expf(a[m] - mx); sum += a[m]; }
  float inv = 1.0f / sum;

  bf16* vr = vec + (size_t)n * 2354;
  const float* cp = cont + (size_t)n * 16 * 768;
  for (int h = threadIdx.x; h < 768; h += 256) {
    float ws = 0.f;
    #pragma unroll
    for (int m = 0; m < 16; ++m) ws += a[m] * cp[m * 768 + h];
    vr[h]        = (bf16)startv[(size_t)n * 768 + h];
    vr[768 + h]  = (bf16)endv[(size_t)n * 768 + h];
    vr[1536 + h] = (bf16)(ws * inv);
  }
  int wi = (w < 4) ? w : 4;
  int t  = tlab[n];
  for (int i = threadIdx.x; i < 50; i += 256) {
    float v = (i < 20) ? wemb[wi * 20 + i] : temb[t * 30 + (i - 20)];
    vr[2304 + i] = (bf16)v;
  }
}

// ---------------------------------------------------------------------------
// g[e][n] = sigmoid(dot(vec[n], gate_W[e]) + gate_b[e]); one wave per (e,n)
// ---------------------------------------------------------------------------
__global__ void gate_kernel(const bf16* __restrict__ vec,
                            const float* __restrict__ gW,
                            const float* __restrict__ gb,
                            float* __restrict__ g)
{
  int gtid = blockIdx.x * 256 + threadIdx.x;
  int wid  = gtid >> 5;
  int lane = threadIdx.x & 31;
  if (wid >= 8 * 2048) return;
  int e = wid >> 11;
  int n = wid & 2047;
  const bf16*  vp = vec + (size_t)n * 2354;
  const float* wp = gW  + (size_t)e * 2354;
  float s = 0.f;
  for (int k = lane; k < 2354; k += 32) s += (float)vp[k] * wp[k];
  #pragma unroll
  for (int off = 16; off > 0; off >>= 1) s += __shfl_down(s, off, 32);
  if (lane == 0) g[e * 2048 + n] = 1.0f / (1.0f + __expf(-(s + gb[e])));
}

// ---------------------------------------------------------------------------
// adjhat[b][s][e][t..t+1] = bf16( adjm[b][e][s][t..t+1] * g[e][b*512+s] )
// permute (B,E,S,S) -> (B,S,E,S); 2 elems/thread: b64 load, b32 store
// ---------------------------------------------------------------------------
__global__ void adjhat_kernel(const float* __restrict__ adjm,
                              const float* __restrict__ g,
                              bf16* __restrict__ ah)
{
  long i2 = (long)blockIdx.x * 256 + threadIdx.x;
  if (i2 >= ((long)4 * 8 * 512 * 512) / 2) return;
  long i = i2 << 1;                       // even element index
  int t = (int)(i & 511); long r = i >> 9;
  int s = (int)(r & 511); r >>= 9;
  int e = (int)(r & 7);   int b = (int)(r >> 3);
  f32x2 av = *(const f32x2*)&adjm[i];
  float gg = g[e * 2048 + b * 512 + s];
  unsigned int u = pack_bf16(av.x * gg, av.y * gg);
  *(unsigned int*)&ah[((((long)b * 512 + s) * 8 + e) << 9) + t] = u;
}

// ---------------------------------------------------------------------------
extern "C" void kernel_launch(void* const* d_in, const int* in_sizes, int n_in,
                              void* d_out, int out_size, void* d_ws, size_t ws_size,
                              hipStream_t stream)
{
  const int B_ = 4, S_ = 512, T_ = 1024, M_ = 16, H_ = 768, E_ = 8,
            HID_ = 128, D_ = 2354;
  (void)in_sizes; (void)n_in; (void)out_size; (void)ws_size;

  const float* doc  = (const float*)d_in[0];
  const float* smap = (const float*)d_in[1];
  const float* emap = (const float*)d_in[2];
  const float* cmap = (const float*)d_in[3];
  const float* adjm = (const float*)d_in[4];
  const float* W1   = (const float*)d_in[5];
  const float* b1   = (const float*)d_in[6];
  const float* W2   = (const float*)d_in[7];
  const float* b2   = (const float*)d_in[8];
  const float* wemb = (const float*)d_in[9];
  const float* temb = (const float*)d_in[10];
  const float* gW   = (const float*)d_in[11];
  const float* gb   = (const float*)d_in[12];
  const float* cW   = (const float*)d_in[13];
  const float* cb   = (const float*)d_in[14];
  const int*   wid  = (const int*)d_in[15];
  const int*   tlb  = (const int*)d_in[16];
  float* out = (float*)d_out;

  // workspace layout (~182 MB)
  char* p = (char*)d_ws;
  float* start_f = (float*)p; p += (size_t)B_ * S_ * H_ * 4;
  float* end_f   = (float*)p; p += (size_t)B_ * S_ * H_ * 4;
  float* cont_f  = (float*)p; p += (size_t)B_ * S_ * M_ * H_ * 4;
  float* h1_f    = (float*)p; p += (size_t)B_ * S_ * M_ * HID_ * 4;
  float* sc_f    = (float*)p; p += (size_t)B_ * S_ * M_ * 4;
  bf16*  vec_b   = (bf16*)p;  p += (size_t)B_ * S_ * D_ * 2;
  float* g_f     = (float*)p; p += (size_t)E_ * B_ * S_ * 4;
  bf16*  adjh_b  = (bf16*)p;  p += (size_t)B_ * S_ * E_ * S_ * 2;
  bf16*  hconv_b = (bf16*)p;  p += (size_t)B_ * E_ * S_ * H_ * 2;

  // 1) start / end = mappings x doc   (per-b 512x1024 * 1024x768)
  {
    dim3 g(S_ / 128, H_ / 128, B_);
    gemm_wmma_bf16<float, float, false, false, float><<<g, 256, 0, stream>>>(
        smap, doc, nullptr, start_f, S_, H_, T_,
        (long)T_, (long)H_, (long)H_, B_,
        (long)S_ * T_, 0, (long)T_ * H_, 0, (long)S_ * H_, 0, 0, 0);
    gemm_wmma_bf16<float, float, false, false, float><<<g, 256, 0, stream>>>(
        emap, doc, nullptr, end_f, S_, H_, T_,
        (long)T_, (long)H_, (long)H_, B_,
        (long)S_ * T_, 0, (long)T_ * H_, 0, (long)S_ * H_, 0, 0, 0);
  }
  // 2) cont = cont_mappings x doc     (per-b 8192x1024 * 1024x768)
  {
    dim3 g(S_ * M_ / 128, H_ / 128, B_);
    gemm_wmma_bf16<float, float, false, false, float><<<g, 256, 0, stream>>>(
        cmap, doc, nullptr, cont_f, S_ * M_, H_, T_,
        (long)T_, (long)H_, (long)H_, B_,
        (long)S_ * M_ * T_, 0, (long)T_ * H_, 0, (long)S_ * M_ * H_, 0, 0, 0);
  }
  // 3) h1 = relu(cont x W1^T + b1)    (32768x768 * 768x128, B transposed)
  {
    dim3 g(B_ * S_ * M_ / 128, HID_ / 128, 1);
    gemm_wmma_bf16<float, float, true, true, float><<<g, 256, 0, stream>>>(
        cont_f, W1, b1, h1_f, B_ * S_ * M_, HID_, H_,
        (long)H_, (long)H_, (long)HID_, 1,
        0, 0, 0, 0, 0, 0, 0, 0);
  }
  // 4) sc = h1 . W2 + b2
  sc_dot_kernel<<<(B_ * S_ * M_ + 255) / 256, 256, 0, stream>>>(
      h1_f, W2, b2, sc_f, B_ * S_ * M_);
  // 5) masked softmax + wsum + span-vector assembly (bf16)
  softmax_vec_kernel<<<B_ * S_, 256, 0, stream>>>(
      sc_f, cont_f, start_f, end_f, wemb, temb, wid, tlb, vec_b);
  // 6) gates
  gate_kernel<<<(E_ * B_ * S_ * 32 + 255) / 256, 256, 0, stream>>>(
      vec_b, gW, gb, g_f);
  // 7) gated+permuted adjacency -> bf16 (2 elements per thread)
  adjhat_kernel<<<(int)(((long)B_ * E_ * S_ * S_) / 512), 256, 0, stream>>>(
      adjm, g_f, adjh_b);
  // 8) hconv = vec x conv_W^T + conv_b  -> (B,E,S,H) bf16; batched over (e,b)
  {
    dim3 g(S_ / 128, H_ / 128, E_ * B_);
    gemm_wmma_bf16<bf16, float, true, false, bf16><<<g, 256, 0, stream>>>(
        vec_b, cW, cb, hconv_b, S_, H_, D_,
        (long)D_, (long)D_, (long)H_, B_,
        (long)S_ * D_, 0,                       // A: stride per b, none per e
        0, (long)H_ * D_,                       // B: none per b, conv_W[e]
        (long)E_ * S_ * H_, (long)S_ * H_,      // C: (B,E,S,H)
        0, (long)H_);                           // bias: conv_b[e]
  }
  // 9) out = relu( adjhat (B,S,4096) x hconv (B,4096,768) )
  {
    dim3 g(S_ / 128, H_ / 128, B_);
    gemm_wmma_bf16<bf16, bf16, false, true, float><<<g, 256, 0, stream>>>(
        adjh_b, hconv_b, nullptr, out, S_, H_, E_ * S_,
        (long)E_ * S_, (long)H_, (long)H_, B_,
        (long)S_ * E_ * S_, 0, (long)E_ * S_ * H_, 0, (long)S_ * H_, 0, 0, 0);
  }
}
